// ToyNet_73589969650227
// MI455X (gfx1250) — 8-bit matrix operand order repaired and checked by reading, compiled, not run on hardware
//
#include <hip/hip_runtime.h>
#include <hip/hip_bf16.h>
#include <stdint.h>


#define KDIM   12288
#define MDIM   8192
#define NDIM   1000
#define NPAD   1024

#define LDS_STRIDE 144
#define ABUF_BYTES (128 * LDS_STRIDE)
#define KSTEPS     (KDIM / 128)

typedef __attribute__((ext_vector_type(16))) int   v16i;
typedef __attribute__((ext_vector_type(4)))  int   v4i;
typedef __attribute__((ext_vector_type(2)))  int   v2i;
typedef __attribute__((ext_vector_type(8)))  float v8f;

__device__ __forceinline__ unsigned char sign_fp8(float x) {
    return x > 0.0f ? 0x38u : (x < 0.0f ? 0xB8u : 0x00u);
}

__device__ __forceinline__ void async_copy16(uint32_t lds_off, const uint8_t* g) {
    asm volatile("global_load_async_to_lds_b128 %0, %1, off"
                 :: "v"(lds_off), "v"(g) : "memory");
}
__device__ __forceinline__ void wait_async0() {
    asm volatile("s_wait_asynccnt 0" ::: "memory");
}

__global__ __launch_bounds__(256) void pack_a_kernel(const float* __restrict__ x,
                                                     uint8_t* __restrict__ y) {
    long i = ((long)blockIdx.x * blockDim.x + threadIdx.x) * 4;
    if (i >= (long)MDIM * KDIM) return;
    const float4 v = *(const float4*)(x + i);
    uint32_t p = (uint32_t)sign_fp8(v.x)
               | ((uint32_t)sign_fp8(v.y) << 8)
               | ((uint32_t)sign_fp8(v.z) << 16)
               | ((uint32_t)sign_fp8(v.w) << 24);
    *(uint32_t*)(y + i) = p;
}

__global__ __launch_bounds__(256) void pack_w_kernel(const float* __restrict__ w,
                                                     uint8_t* __restrict__ y) {
    long i = ((long)blockIdx.x * blockDim.x + threadIdx.x) * 4;
    if (i >= (long)NPAD * KDIM) return;
    long row = i / KDIM;
    uint32_t p = 0u;
    if (row < NDIM) {
        const float4 v = *(const float4*)(w + i);
        p = (uint32_t)sign_fp8(v.x)
          | ((uint32_t)sign_fp8(v.y) << 8)
          | ((uint32_t)sign_fp8(v.z) << 16)
          | ((uint32_t)sign_fp8(v.w) << 24);
    }
    *(uint32_t*)(y + i) = p;
}

__global__ __launch_bounds__(256) void bgemm_fp8_kernel(const uint8_t* __restrict__ Ap,
                                                        const uint8_t* __restrict__ Bp,
                                                        float* __restrict__ out) {
    extern __shared__ uint8_t smem[];

    const int lane  = threadIdx.x & 31;
    const int wave  = threadIdx.x >> 5;
    const int waveM = wave & 1;
    const int waveN = wave >> 1;
    const int lo    = lane & 15;
    const int hi    = lane >> 4;

    const int mBase = blockIdx.y * 128;
    const int nBase = blockIdx.x * 128;

    const int c16 = (threadIdx.x & 7) * 16;
    const int r0  = threadIdx.x >> 3;
    const uint8_t* asrc = Ap + (long)(mBase + r0) * KDIM + c16;
    const uint32_t adst = (uint32_t)(r0 * LDS_STRIDE + c16);

    const uint8_t* brow0 = Bp + (long)(nBase + waveN * 32 + lo) * KDIM + hi * 16;
    const uint8_t* brow1 = brow0 + (long)16 * KDIM;

    const uint32_t aFragBase = (uint32_t)((waveM * 64 + lo) * LDS_STRIDE + hi * 16);

    v8f acc[4][2];
#pragma unroll
    for (int s = 0; s < 4; ++s) {
        acc[s][0] = (v8f){};
        acc[s][1] = (v8f){};
    }

#pragma unroll
    for (int i = 0; i < 4; ++i)
        async_copy16(adst + i * 32 * LDS_STRIDE, asrc + (long)i * 32 * KDIM);

    for (int k = 0; k < KSTEPS; ++k) {
        const int k0 = k * 128;
        const uint32_t bufCur = (uint32_t)(k & 1) * ABUF_BYTES;
        const uint32_t bufNxt = (uint32_t)(~k & 1) * ABUF_BYTES;

        wait_async0();
        __syncthreads();

        if (k + 1 < KSTEPS) {
#pragma unroll
            for (int i = 0; i < 4; ++i)
                async_copy16(bufNxt + adst + i * 32 * LDS_STRIDE,
                             asrc + (k0 + 128) + (long)i * 32 * KDIM);
            __builtin_prefetch(brow0 + k0 + 128, 0, 1);
        }

        v16i bf0, bf1;
#pragma unroll
        for (int g = 0; g < 4; ++g) {
            v4i t0 = *(const v4i*)(brow0 + k0 + g * 32);
            v4i t1 = *(const v4i*)(brow1 + k0 + g * 32);
            bf0[4 * g + 0] = t0[0]; bf0[4 * g + 1] = t0[1];
            bf0[4 * g + 2] = t0[2]; bf0[4 * g + 3] = t0[3];
            bf1[4 * g + 0] = t1[0]; bf1[4 * g + 1] = t1[1];
            bf1[4 * g + 2] = t1[2]; bf1[4 * g + 3] = t1[3];
        }

#pragma unroll
        for (int s = 0; s < 4; ++s) {
            const uint32_t abase = bufCur + aFragBase + (uint32_t)(s * 16 * LDS_STRIDE);
            v16i af;
#pragma unroll
            for (int p = 0; p < 8; ++p) {
                v2i t = *(const v2i*)(smem + abase + (p >> 1) * 32 + (p & 1) * 8);
                af[2 * p + 0] = t[0];
                af[2 * p + 1] = t[1];
            }
            acc[s][0] = __builtin_amdgcn_wmma_f32_16x16x128_fp8_fp8(
                af, bf0, (short)0, acc[s][0], false, false);
            acc[s][1] = __builtin_amdgcn_wmma_f32_16x16x128_fp8_fp8(
                af, bf1, (short)0, acc[s][1], false, false);
        }

        __syncthreads();
    }

#pragma unroll
    for (int x = 0; x < 2; ++x) {
        const int col = nBase + waveN * 32 + x * 16 + lo;
        if (col < NDIM) {
#pragma unroll
            for (int s = 0; s < 4; ++s) {
#pragma unroll
                for (int r = 0; r < 8; ++r) {
                    const int row = mBase + waveM * 64 + s * 16 + hi * 8 + r;
                    out[(long)row * NDIM + col] = acc[s][x][r];
                }
            }
        }
    }
}

extern "C" void kernel_launch(void* const* d_in, const int* in_sizes, int n_in,
                              void* d_out, int out_size, void* d_ws, size_t ws_size,
                              hipStream_t stream) {
    const float* x = (const float*)d_in[0];
    const float* w = (const float*)d_in[1];
    float* out = (float*)d_out;

    uint8_t* Ap = (uint8_t*)d_ws;
    uint8_t* Bp = Ap + (size_t)MDIM * KDIM;
    (void)in_sizes; (void)n_in; (void)out_size; (void)ws_size;

    {
        long n4 = (long)MDIM * KDIM / 4;
        pack_a_kernel<<<dim3((unsigned)((n4 + 255) / 256)), 256, 0, stream>>>(x, Ap);
    }
    {
        long n4 = (long)NPAD * KDIM / 4;
        pack_w_kernel<<<dim3((unsigned)((n4 + 255) / 256)), 256, 0, stream>>>(w, Bp);
    }
    {
        dim3 grid(NPAD / 128, MDIM / 128);
        bgemm_fp8_kernel<<<grid, 256, 2 * ABUF_BYTES, stream>>>(Ap, Bp, out);
    }
}
